// MultiHeadDistanceLayer_18133351924245
// MI455X (gfx1250) — matrix operand order repaired and checked by reading, not compiled, not run on hardware
//
#include <hip/hip_runtime.h>
#include <hip/hip_bf16.h>

typedef __attribute__((ext_vector_type(16))) _Float16 v16h;
typedef __attribute__((ext_vector_type(8)))  _Float16 v8h;
typedef __attribute__((ext_vector_type(8)))  float    v8f;

#define B_   4
#define L_   2048
#define D_   256
#define H_   8
#define HD_  64
#define NH_  (H_ * HD_)          // 512
#define INV_MAXLEN (1.0f / 2048.0f)

#define SHUF16(lo, hi) __builtin_shufflevector(lo, hi, 0,1,2,3,4,5,6,7,8,9,10,11,12,13,14,15)

// ---------------- Stage 1: precision prep ----------------

__global__ __launch_bounds__(256) void cvt_x_f16(const float* __restrict__ x,
                                                 _Float16* __restrict__ xh, int n) {
  int i = blockIdx.x * 256 + threadIdx.x;
  if (i < n) xh[i] = (_Float16)x[i];
}

// W: [D_, NH_] f32  ->  Wt: [NH_, D_] f16 (transposed so WMMA-B loads are contiguous)
__global__ __launch_bounds__(256) void transpose_w_f16(const float* __restrict__ W,
                                                       _Float16* __restrict__ Wt) {
  int o = blockIdx.x * 256 + threadIdx.x;          // o < NH_*D_
  int n = o >> 8;                                   // /D_
  int k = o & (D_ - 1);
  Wt[o] = (_Float16)W[k * NH_ + n];
}

// ---------------- Stage 2: Q/K projection GEMM (WMMA f16) ----------------
// One wave per 16x16 output tile. M = B_*L_ = 8192, N = NH_ = 512, K = D_ = 256.
__global__ __launch_bounds__(256) void qk_proj(const _Float16* __restrict__ xh,
                                               const _Float16* __restrict__ Wqt,
                                               const _Float16* __restrict__ Wkt,
                                               const float* __restrict__ bq,
                                               const float* __restrict__ bk,
                                               _Float16* __restrict__ Qh,
                                               _Float16* __restrict__ Kh) {
  const int lane = threadIdx.x & 31;
  const int wid  = (blockIdx.x * 256 + threadIdx.x) >> 5;  // 0..32767
  const int nt   = wid & 31;                 // 32 n-tiles
  const int mt   = (wid >> 5) & 511;         // 512 m-tiles
  const int mat  = wid >> 14;                // 0 = Q, 1 = K

  const _Float16* Wt   = mat ? Wkt : Wqt;
  const float*    bias = mat ? bk  : bq;
  _Float16*       Out  = mat ? Kh  : Qh;

  const int hf  = lane >> 4;    // half-wave id
  const int l16 = lane & 15;

  const _Float16* aptr = xh + (mt * 16 + l16) * D_;   // row of x (f16)
  const _Float16* bptr = Wt + (nt * 16 + l16) * D_;   // column of W (row of Wt)

  v8f acc = {};
#pragma unroll
  for (int kk = 0; kk < D_; kk += 32) {
    // A 16x32 f16: lane<16 holds K {kk..kk+7, kk+16..kk+23}; lane>=16 shifted by 8
    const int aoff = kk + hf * 8;
    v8h alo = *(const v8h*)(aptr + aoff);
    v8h ahi = *(const v8h*)(aptr + aoff + 16);
    v16h A = SHUF16(alo, ahi);
    // B 32x16 f16: lane<16 holds K {kk..kk+7, kk+16..kk+23} of its column; lane>=16 shifted by 8
    const int boff = kk + hf * 8;
    v8h blo = *(const v8h*)(bptr + boff);
    v8h bhi = *(const v8h*)(bptr + boff + 16);
    v16h Bm = SHUF16(blo, bhi);
    acc = __builtin_amdgcn_wmma_f32_16x16x32_f16(false, A, false, Bm,
                                                 (short)0, acc, false, false);
  }

  const int col = nt * 16 + l16;        // N index owned by this lane
  const float bv = bias[col];           // bq/bk flattened == column index
#pragma unroll
  for (int v = 0; v < 8; ++v) {
    int row = mt * 16 + v + 8 * hf;     // C layout: vgpr v -> M = v (+8 for upper half)
    Out[row * NH_ + col] = (_Float16)(acc[v] + bv);
  }
}

// ---------------- Stage 3: prior-weighted attention -> distance ----------------
// One wave per (h, b, 16-query tile); streams all 2048 keys in 16-key WMMA tiles.
__global__ __launch_bounds__(256) void attn_dist(const _Float16* __restrict__ Qh,
                                                 const _Float16* __restrict__ Kh,
                                                 const float* __restrict__ prior_mean,
                                                 const float* __restrict__ log_prior_std,
                                                 float* __restrict__ out) {
  const float INV_SQRT_2PI = 0.3989422804014327f;
  const int lane = threadIdx.x & 31;
  const int wid  = (blockIdx.x * 256 + threadIdx.x) >> 5;  // 0..4095
  const int qt   = wid & 127;          // 128 query tiles
  const int hb   = wid >> 7;           // 32 (h,b) pairs
  const int h    = hb & (H_ - 1);
  const int b    = hb >> 3;

  const int hf  = lane >> 4;
  const int l16 = lane & 15;

  const float mean = prior_mean[h];
  const float istd = __expf(-log_prior_std[h]);       // 1/std
  const float coef = INV_SQRT_2PI * istd;

  const int qbase = qt * 16;
  const _Float16* qptr = Qh + (size_t)(b * L_ + qbase + l16) * NH_ + h * HD_;

  // Loop-invariant A operands for hd 0..31 and 32..63
  const int aoff = hf * 8;
  v16h A0 = SHUF16(*(const v8h*)(qptr + aoff),      *(const v8h*)(qptr + aoff + 16));
  v16h A1 = SHUF16(*(const v8h*)(qptr + 32 + aoff), *(const v8h*)(qptr + 32 + aoff + 16));

  // Per-lane online softmax state: lane owns score column n = l16 of each row v
  float mrow[8], den[8], num[8];
#pragma unroll
  for (int v = 0; v < 8; ++v) { mrow[v] = -__builtin_inff(); den[v] = 0.f; num[v] = 0.f; }

  const _Float16* kslice = Kh + (size_t)(b * L_) * NH_ + h * HD_;
  const int boff = hf * 8;

  for (int kt = 0; kt < L_; kt += 16) {
    const _Float16* kptr = kslice + (size_t)(kt + l16) * NH_;
    v16h B0 = SHUF16(*(const v8h*)(kptr + boff),      *(const v8h*)(kptr + boff + 16));
    v16h B1 = SHUF16(*(const v8h*)(kptr + 32 + boff), *(const v8h*)(kptr + 32 + boff + 16));
    v8f c = {};
    c = __builtin_amdgcn_wmma_f32_16x16x32_f16(false, A0, false, B0, (short)0, c, false, false);
    c = __builtin_amdgcn_wmma_f32_16x16x32_f16(false, A1, false, B1, (short)0, c, false, false);

    const float kidx = (float)(kt + l16);
#pragma unroll
    for (int v = 0; v < 8; ++v) {
      float qidx = (float)(qbase + v + 8 * hf);
      float dist = (kidx - qidx) * INV_MAXLEN;
      float u    = (dist - mean) * istd;
      float g    = coef * __expf(-0.5f * u * u);      // gaussian prior
      float s    = c[v] * 0.125f * g;                 // (qk / sqrt(64)) * prior
      float mo   = mrow[v];
      float mn   = fmaxf(mo, s);
      float al   = __expf(mo - mn);
      float e    = __expf(s - mn);
      den[v] = den[v] * al + e;
      num[v] = num[v] * al + e * dist;
      mrow[v] = mn;
    }
  }

  // Merge 16 per-lane partial softmaxes per row (stays within each 16-lane half)
#pragma unroll
  for (int v = 0; v < 8; ++v) {
    float m = mrow[v], dn = den[v], nm = num[v];
#pragma unroll
    for (int off = 1; off < 16; off <<= 1) {
      float m2 = __shfl_xor(m, off, 32);
      float d2 = __shfl_xor(dn, off, 32);
      float n2 = __shfl_xor(nm, off, 32);
      float mn = fmaxf(m, m2);
      float a1 = __expf(m - mn);
      float a2 = __expf(m2 - mn);
      dn = dn * a1 + d2 * a2;
      nm = nm * a1 + n2 * a2;
      m = mn;
    }
    if (l16 == 0) {
      int qr = qbase + v + 8 * hf;
      out[(size_t)(b * L_ + qr) * H_ + h] = nm / dn;   // [b, l, h]
    }
  }
}

// ---------------- host-side launch ----------------

extern "C" void kernel_launch(void* const* d_in, const int* in_sizes, int n_in,
                              void* d_out, int out_size, void* d_ws, size_t ws_size,
                              hipStream_t stream) {
  const float* x   = (const float*)d_in[0];   // [B, L, D]
  const float* Wq  = (const float*)d_in[1];   // [D, H*HD]
  const float* bq  = (const float*)d_in[2];   // [H,1,1,HD] -> flat 512
  const float* Wk  = (const float*)d_in[3];
  const float* bk  = (const float*)d_in[4];
  const float* pm  = (const float*)d_in[5];   // prior_mean [H]
  const float* lps = (const float*)d_in[6];   // log_prior_std [H]
  float* out = (float*)d_out;                 // [B, L, H] f32

  // Workspace layout (f16), all 16-byte aligned
  char* ws = (char*)d_ws;
  _Float16* xh  = (_Float16*)ws;                               // 8192*256
  _Float16* Wqt = (_Float16*)(ws + (size_t)B_*L_*D_*2);        // 512*256
  _Float16* Wkt = Wqt + NH_ * D_;
  _Float16* Qh  = Wkt + NH_ * D_;                              // 8192*512
  _Float16* Kh  = Qh + (size_t)B_ * L_ * NH_;

  const int nx = B_ * L_ * D_;                                 // 2,097,152
  cvt_x_f16<<<nx / 256, 256, 0, stream>>>(x, xh, nx);
  transpose_w_f16<<<(NH_ * D_) / 256, 256, 0, stream>>>(Wq, Wqt);
  transpose_w_f16<<<(NH_ * D_) / 256, 256, 0, stream>>>(Wk, Wkt);

  // 2 matrices * 512 m-tiles * 32 n-tiles = 32768 waves, 8 waves/block
  qk_proj<<<4096, 256, 0, stream>>>(xh, Wqt, Wkt, bq, bk, Qh, Kh);

  // 32 (h,b) * 128 q-tiles = 4096 waves, 8 waves/block
  attn_dist<<<512, 256, 0, stream>>>(Qh, Kh, pm, lps, out);
}
